// DisNets_83580063580403
// MI455X (gfx1250) — matrix operand order repaired and checked by reading, not compiled, not run on hardware
//
#include <hip/hip_runtime.h>
#include <hip/hip_bf16.h>

typedef __attribute__((ext_vector_type(16))) _Float16 v16h;
typedef __attribute__((ext_vector_type(8)))  _Float16 h8;
typedef __attribute__((ext_vector_type(8)))  float    v8f;

// ---------------------------------------------------------------------------
// K1a: deg only.  s[i] = rsqrt(1 + sum_j adj[i][j])
// ---------------------------------------------------------------------------
__global__ __launch_bounds__(256)
void degree_rsqrt(const float* __restrict__ adj, float* __restrict__ s, int n) {
    __shared__ float red[256];
    const float4* row = (const float4*)(adj + (size_t)blockIdx.x * (size_t)n);
    float sum = 0.0f;
    for (int t = threadIdx.x; t < (n >> 2); t += 256) {
        float4 v = row[t];
        sum += v.x + v.y + v.z + v.w;
    }
    red[threadIdx.x] = sum;
    __syncthreads();
    for (int st = 128; st > 0; st >>= 1) {
        if (threadIdx.x < st) red[threadIdx.x] += red[threadIdx.x + st];
        __syncthreads();
    }
    if (threadIdx.x == 0) s[blockIdx.x] = rsqrtf(red[0] + 1.0f);
}

// ---------------------------------------------------------------------------
// K1b: fused degree + lossless f32->f16 conversion of A (entries are 0/1).
// Reads 1.07 GB once, writes 0.54 GB; the 3 SpMM layers then stream f16.
// ---------------------------------------------------------------------------
__global__ __launch_bounds__(256)
void degree_rsqrt_cvt(const float* __restrict__ adj, _Float16* __restrict__ a16,
                      float* __restrict__ s, int n) {
    __shared__ float red[256];
    const float4* row = (const float4*)(adj + (size_t)blockIdx.x * (size_t)n);
    h8* out = (h8*)(a16 + (size_t)blockIdx.x * (size_t)n);
    float sum = 0.0f;
    for (int t = threadIdx.x; t < (n >> 3); t += 256) {
        float4 v0 = row[2 * t], v1 = row[2 * t + 1];
        sum += v0.x + v0.y + v0.z + v0.w + v1.x + v1.y + v1.z + v1.w;
        h8 h;
        h[0] = (_Float16)v0.x; h[1] = (_Float16)v0.y;
        h[2] = (_Float16)v0.z; h[3] = (_Float16)v0.w;
        h[4] = (_Float16)v1.x; h[5] = (_Float16)v1.y;
        h[6] = (_Float16)v1.z; h[7] = (_Float16)v1.w;
        out[t] = h;
    }
    red[threadIdx.x] = sum;
    __syncthreads();
    for (int st = 128; st > 0; st >>= 1) {
        if (threadIdx.x < st) red[threadIdx.x] += red[threadIdx.x + st];
        __syncthreads();
    }
    if (threadIdx.x == 0) s[blockIdx.x] = rsqrtf(red[0] + 1.0f);
}

// ---------------------------------------------------------------------------
// K2: Y = h_prev @ W, write transposed+scaled f16: ypT[j][i] = s[i]*Y[i][j]
// ---------------------------------------------------------------------------
__global__ __launch_bounds__(256)
void feat_xw_scale(const float* __restrict__ hprev, const float* __restrict__ W,
                   const float* __restrict__ s, _Float16* __restrict__ ypT,
                   int n, int din, int dout) {
    extern __shared__ float wlds[];
    for (int t = threadIdx.x; t < din * dout; t += 256) wlds[t] = W[t];
    __syncthreads();
    int idx = blockIdx.x * 256 + threadIdx.x;
    int i = idx / dout;
    int j = idx - i * dout;
    const float* hrow = hprev + (size_t)i * din;
    float acc = 0.0f;
    for (int k = 0; k < din; ++k) acc += hrow[k] * wlds[k * dout + j];
    ypT[(size_t)j * n + i] = (_Float16)(s[i] * acc);
}

// ---------------------------------------------------------------------------
// helper: one K-step: NT B-fragment loads + NT WMMAs against a shared A frag
// ---------------------------------------------------------------------------
template<int NT>
__device__ __forceinline__
void wmma_step(const v16h& a, const _Float16* __restrict__ ypT, int n, int kb,
               int lane16, int hi, v8f (&acc)[NT]) {
#pragma unroll
    for (int t = 0; t < NT; ++t) {
        const _Float16* bptr =
            ypT + (size_t)(t * 16 + lane16) * (size_t)n + kb + hi * 8;
        h8 blo = *(const h8*)(bptr);
        h8 bhi = *(const h8*)(bptr + 16);
        v16h b;
#pragma unroll
        for (int e = 0; e < 8; ++e) { b[e] = blo[e]; b[e + 8] = bhi[e]; }
        acc[t] = __builtin_amdgcn_wmma_f32_16x16x32_f16(
            false, a, false, b, (short)0, acc[t], false, false);
    }
}

// ---------------------------------------------------------------------------
// K3: Z = A @ Y' via v_wmma_f32_16x16x32_f16 with split-K x4:
//   block = 128 threads = 4 waves, ALL on the same 16-row strip of A;
//   wave w sweeps K in [w*n/4, (w+1)*n/4), partials reduced through LDS.
//   -> 4096 independent waves / 1024 blocks for MLP-level parallelism,
//      while A is still read exactly once per layer.
// Epilogue: h_next[i][j] = relu( s[i]*(Z[i][j] + Y'[i][j]) + b[j] )
// F16A=true streams the pre-converted f16 adjacency (half the bytes, no cvt).
// ---------------------------------------------------------------------------
template<int NT, bool F16A>
__global__ __launch_bounds__(128)
void gcn_spmm_wmma(const void* __restrict__ adjv, const _Float16* __restrict__ ypT,
                   const float* __restrict__ s, const float* __restrict__ bias,
                   float* __restrict__ hnext, int n, int ncols) {
    const int wave   = threadIdx.x >> 5;
    const int lane   = threadIdx.x & 31;
    const int lane16 = lane & 15;
    const int hi     = lane >> 4;
    const int m0     = blockIdx.x * 16;
    const int row    = m0 + lane16;
    const int kslice = n >> 2;
    const int k0     = wave * kslice;
    const int k1     = k0 + kslice;

    v8f acc[NT];
#pragma unroll
    for (int t = 0; t < NT; ++t)
        acc[t] = (v8f){0.f, 0.f, 0.f, 0.f, 0.f, 0.f, 0.f, 0.f};

    if (F16A) {
        const _Float16* arow = (const _Float16*)adjv + (size_t)row * (size_t)n;
#pragma unroll 2
        for (int kb = k0; kb < k1; kb += 32) {
            __builtin_prefetch(arow + kb + 2048, 0, 0);
            const int ka = kb + hi * 8;
            h8 a0 = *(const h8*)(arow + ka);
            h8 a1 = *(const h8*)(arow + ka + 16);
            v16h a;
#pragma unroll
            for (int e = 0; e < 8; ++e) { a[e] = a0[e]; a[e + 8] = a1[e]; }
            wmma_step<NT>(a, ypT, n, kb, lane16, hi, acc);
        }
    } else {
        const float* arow = (const float*)adjv + (size_t)row * (size_t)n;
#pragma unroll 2
        for (int kb = k0; kb < k1; kb += 32) {
            __builtin_prefetch(arow + kb + 2048, 0, 0);
            const int ka = kb + hi * 8;
            float4 f0 = *(const float4*)(arow + ka);
            float4 f1 = *(const float4*)(arow + ka + 4);
            float4 f2 = *(const float4*)(arow + ka + 16);
            float4 f3 = *(const float4*)(arow + ka + 20);
            v16h a;
            a[0]  = (_Float16)f0.x; a[1]  = (_Float16)f0.y;
            a[2]  = (_Float16)f0.z; a[3]  = (_Float16)f0.w;
            a[4]  = (_Float16)f1.x; a[5]  = (_Float16)f1.y;
            a[6]  = (_Float16)f1.z; a[7]  = (_Float16)f1.w;
            a[8]  = (_Float16)f2.x; a[9]  = (_Float16)f2.y;
            a[10] = (_Float16)f2.z; a[11] = (_Float16)f2.w;
            a[12] = (_Float16)f3.x; a[13] = (_Float16)f3.y;
            a[14] = (_Float16)f3.z; a[15] = (_Float16)f3.w;
            wmma_step<NT>(a, ypT, n, kb, lane16, hi, acc);
        }
    }

    // ---- split-K reduction through LDS (waves 1..3 -> wave 0) ----
    __shared__ float red[3][NT * 256];
    if (wave != 0) {
#pragma unroll
        for (int t = 0; t < NT; ++t)
#pragma unroll
            for (int r = 0; r < 8; ++r)
                red[wave - 1][t * 256 + r * 32 + lane] = acc[t][r];
    }
    __syncthreads();
    if (wave == 0) {
#pragma unroll
        for (int w = 0; w < 3; ++w)
#pragma unroll
            for (int t = 0; t < NT; ++t)
#pragma unroll
                for (int r = 0; r < 8; ++r)
                    acc[t][r] += red[w][t * 256 + r * 32 + lane];

        // ---- epilogue: C/D layout VGPR r -> M = r + hi*8, N = lane16 ----
#pragma unroll
        for (int t = 0; t < NT; ++t) {
            const int ncol = t * 16 + lane16;
            const float bj = bias[ncol];
#pragma unroll
            for (int r = 0; r < 8; ++r) {
                const int m = m0 + hi * 8 + r;
                const float yp = (float)ypT[(size_t)ncol * n + m];
                float v = s[m] * (acc[t][r] + yp) + bj;
                hnext[(size_t)m * ncols + ncol] = v > 0.f ? v : 0.f;
            }
        }
    }
}

// ---------------------------------------------------------------------------
// K4a: partial column sums of h3 (N x 64) over 128-row slabs
// ---------------------------------------------------------------------------
__global__ __launch_bounds__(256)
void col_partial(const float* __restrict__ h, float* __restrict__ partial, int n) {
    __shared__ float red[256];
    const int col = threadIdx.x & 63;
    const int sub = threadIdx.x >> 6;
    const int r0  = blockIdx.x * 128;
    float sum = 0.f;
    for (int r = sub; r < 128; r += 4) sum += h[(size_t)(r0 + r) * 64 + col];
    red[threadIdx.x] = sum;
    __syncthreads();
    if (threadIdx.x < 128) red[threadIdx.x] += red[threadIdx.x + 128];
    __syncthreads();
    if (threadIdx.x < 64)
        partial[blockIdx.x * 64 + threadIdx.x] = red[threadIdx.x] + red[threadIdx.x + 64];
}

// ---------------------------------------------------------------------------
// K4b: g = mean; h1 = elu(g@Wh1+bh1); logits = h1@Wh2+bh2; softmax.
// ---------------------------------------------------------------------------
__global__ __launch_bounds__(64)
void mlp_head(const float* __restrict__ partial,
              const float* __restrict__ Wh1, const float* __restrict__ bh1,
              const float* __restrict__ Wh2, const float* __restrict__ bh2,
              float* __restrict__ out) {
    __shared__ float g[64];
    __shared__ float h1s[32];
    __shared__ float lg[2];
    const int t = threadIdx.x;
    float sum = 0.f;
    for (int p = 0; p < 128; ++p) sum += partial[p * 64 + t];
    g[t] = sum * (1.0f / 16384.0f);
    __syncthreads();
    if (t < 32) {
        float a = bh1[t];
        for (int k = 0; k < 64; ++k) a += g[k] * Wh1[k * 32 + t];
        h1s[t] = a > 0.f ? a : (expf(a) - 1.0f);
    }
    __syncthreads();
    if (t < 2) {
        float a = bh2[t];
        for (int k = 0; k < 32; ++k) a += h1s[k] * Wh2[k * 2 + t];
        lg[t] = a;
    }
    __syncthreads();
    if (t == 0) {
        float m = fmaxf(lg[0], lg[1]);
        float e0 = expf(lg[0] - m), e1 = expf(lg[1] - m);
        float inv = 1.0f / (e0 + e1);
        out[0] = lg[0]; out[1] = lg[1];
        out[2] = e0 * inv; out[3] = e1 * inv;
    }
}

// ---------------------------------------------------------------------------
extern "C" void kernel_launch(void* const* d_in, const int* in_sizes, int n_in,
                              void* d_out, int out_size, void* d_ws, size_t ws_size,
                              hipStream_t stream) {
    const float* node = (const float*)d_in[0];
    const float* adj  = (const float*)d_in[1];
    const float* W0   = (const float*)d_in[2];
    const float* b0   = (const float*)d_in[3];
    const float* W1   = (const float*)d_in[4];
    const float* b1   = (const float*)d_in[5];
    const float* W2   = (const float*)d_in[6];
    const float* b2   = (const float*)d_in[7];
    const float* Wh1  = (const float*)d_in[8];
    const float* bh1  = (const float*)d_in[9];
    const float* Wh2  = (const float*)d_in[10];
    const float* bh2  = (const float*)d_in[11];

    const int N = 16384;

    // workspace layout:
    //   s       : N f32                 @ 0
    //   partial : 128*64 f32            @ 65536
    //   hA      : N*64 f32              @ 131072
    //   hB      : N*64 f32              @ 131072 + 4 MB
    //   ypT     : 64*N f16              @ 131072 + 8 MB
    //   A16     : N*N f16 (512 MB)      @ 16 MB     [only if ws_size allows]
    char* ws = (char*)d_ws;
    float*    s       = (float*)(ws);
    float*    partial = (float*)(ws + 65536);
    float*    hA      = (float*)(ws + 131072);
    float*    hB      = (float*)(ws + 131072 + 4194304);
    _Float16* ypT     = (_Float16*)(ws + 131072 + 8388608);
    _Float16* A16     = (_Float16*)(ws + 16777216);

    const size_t needA16 = 16777216ull + (size_t)N * (size_t)N * 2ull;
    const bool f16a = (ws_size >= needA16);   // ws_size is fixed -> deterministic

    if (f16a) degree_rsqrt_cvt<<<N, 256, 0, stream>>>(adj, A16, s, N);
    else      degree_rsqrt<<<N, 256, 0, stream>>>(adj, s, N);

    // layer 0: node(64) -> hA(32)
    feat_xw_scale<<<(N * 32) / 256, 256, 64 * 32 * 4, stream>>>(node, W0, s, ypT, N, 64, 32);
    if (f16a) gcn_spmm_wmma<2, true ><<<N / 16, 128, 0, stream>>>(A16, ypT, s, b0, hA, N, 32);
    else      gcn_spmm_wmma<2, false><<<N / 16, 128, 0, stream>>>(adj, ypT, s, b0, hA, N, 32);

    // layer 1: hA(32) -> hB(48)
    feat_xw_scale<<<(N * 48) / 256, 256, 32 * 48 * 4, stream>>>(hA, W1, s, ypT, N, 32, 48);
    if (f16a) gcn_spmm_wmma<3, true ><<<N / 16, 128, 0, stream>>>(A16, ypT, s, b1, hB, N, 48);
    else      gcn_spmm_wmma<3, false><<<N / 16, 128, 0, stream>>>(adj, ypT, s, b1, hB, N, 48);

    // layer 2: hB(48) -> hA(64)
    feat_xw_scale<<<(N * 64) / 256, 256, 48 * 64 * 4, stream>>>(hB, W2, s, ypT, N, 48, 64);
    if (f16a) gcn_spmm_wmma<4, true ><<<N / 16, 128, 0, stream>>>(A16, ypT, s, b2, hA, N, 64);
    else      gcn_spmm_wmma<4, false><<<N / 16, 128, 0, stream>>>(adj, ypT, s, b2, hA, N, 64);

    // graph mean + MLP head
    col_partial<<<128, 256, 0, stream>>>(hA, partial, N);
    mlp_head<<<1, 64, 0, stream>>>(partial, Wh1, bh1, Wh2, bh2, (float*)d_out);
}
